// TGN_8881992368207
// MI455X (gfx1250) — matrix operand order repaired and checked by reading, not compiled, not run on hardware
//
#include <hip/hip_runtime.h>
#include <hip/hip_bf16.h>
#include <stdint.h>

// ---------------------------------------------------------------------------
// Problem constants (from the reference)
// ---------------------------------------------------------------------------
#define NNODE   100000
#define DMDIM   256
#define DEDIM   172
#define DTDIM   100
#define DEMBD   256
#define UU      50000
#define NBATCH  6000
#define KNBR    20
#define MSGD    784          // 2*DM + DE + DT
#define MSGP    800          // MSGD padded to 32
#define G3      768          // 3*DM
#define KIN     528          // DM + DE + DT
#define KINP    544          // KIN padded to 32
#define QIND    356          // DM + DT
#define QINP    384          // QIND padded to 32

// WMMA tiling
#define NT_G    48           // 768/16 output tiles for GRU gates
#define KT_IH   25           // 800/32
#define KT_HH   8            // 256/32
#define NT_256  16           // 256/16
#define KT_KV   17           // 544/32
#define KT_Q    12           // 384/32
#define KT_256  8            // 256/32

typedef __attribute__((ext_vector_type(16))) __bf16 v16bf;
typedef __attribute__((ext_vector_type(8)))  float  v8f;

union FragU {
    uint32_t u[8];
    uint4    q[2];
    v16bf    bf;
};

__device__ __forceinline__ uint32_t f2bf(float a) {
    uint32_t u = __float_as_uint(a);
    u += 0x7FFFu + ((u >> 16) & 1u);   // round-to-nearest-even
    return u >> 16;
}
__device__ __forceinline__ uint32_t pack2bf(float lo, float hi) {
    return f2bf(lo) | (f2bf(hi) << 16);
}
__device__ __forceinline__ v8f v8f_zero() {
    v8f z;
    #pragma unroll
    for (int i = 0; i < 8; ++i) z[i] = 0.0f;
    return z;
}

// B fragment (32x16) from the packed bf16 weight buffer.
// Pack layout: frag_idx*256 dwords, within fragment dword = lane*8 + v.
__device__ __forceinline__ void load_bfrag(const uint4* __restrict__ pack,
                                           int frag_idx, int lane, FragU& f) {
    const uint4* p = pack + (size_t)frag_idx * 64 + (size_t)lane * 2;
    f.q[0] = p[0];
    f.q[1] = p[1];
}

// A fragment (16x32 bf16) from a row-major bf16 tile staged in LDS
// (row stride = `stride` elements). 16-bit A layout: lane half = lane>>4;
// V0..3 hold K = half*8 + {0..7}, V4..7 hold K = 16 + half*8 + {0..7}.
__device__ __forceinline__ void load_afrag_lds(const __bf16* base, int stride,
                                               int mrow, int kk, int lane, FragU& f) {
    const int m    = mrow + (lane & 15);
    const int half = lane >> 4;
    const __bf16* r = base + (size_t)m * stride;
    f.q[0] = *(const uint4*)(r + kk + half * 8);        // ds_load_b128
    f.q[1] = *(const uint4*)(r + kk + 16 + half * 8);   // ds_load_b128
}

// ---------------------------------------------------------------------------
// Inverse-id table (maps node id -> row in h_ws, or -1)
// ---------------------------------------------------------------------------
__global__ void init_inv_kernel(int* __restrict__ inv, int n) {
    int i = blockIdx.x * blockDim.x + threadIdx.x;
    if (i < n) inv[i] = -1;
}
__global__ void fill_inv_kernel(int* __restrict__ inv,
                                const int* __restrict__ unique_nids, int u) {
    int i = blockIdx.x * blockDim.x + threadIdx.x;
    if (i < u) inv[unique_nids[i]] = i;
}

// ---------------------------------------------------------------------------
// Pack cols [col0, col0+Kin) of a row-major fp32 weight W[Nout][ld] into bf16
// WMMA B-fragments. B 32x16 layout: lane n = lane&15; K base = (lane>>4)*8;
// VGPR v holds K = base+2(v&3)+16*(v>>2) (lo), +1 (hi). Zero-padded beyond Kin/Nout.
// ---------------------------------------------------------------------------
__global__ void pack_wmma_b(const float* __restrict__ W, uint32_t* __restrict__ out,
                            int Nout, int Kin, int ld, int col0, int NT, int KT) {
    const int idx   = blockIdx.x * blockDim.x + threadIdx.x;
    const int total = NT * KT * 256;
    if (idx >= total) return;
    const int frag   = idx >> 8;
    const int within = idx & 255;
    const int lane   = within >> 3;
    const int v      = within & 7;
    const int ntile  = frag / KT;
    const int ktile  = frag - ntile * KT;
    const int n  = ntile * 16 + (lane & 15);
    const int kb = ktile * 32 + ((lane >> 4) << 3);
    const int k0 = kb + 2 * (v & 3) + ((v >> 2) << 4);
    const int k1 = k0 + 1;
    const float lo = (n < Nout && k0 < Kin) ? W[(size_t)n * ld + col0 + k0] : 0.0f;
    const float hi = (n < Nout && k1 < Kin) ? W[(size_t)n * ld + col0 + k1] : 0.0f;
    out[idx] = pack2bf(lo, hi);
}

// ---------------------------------------------------------------------------
// Generic batched WMMA GEMM, N fixed to 256 (16 tiles, 2 per wave):
//   C[M,256] = act( A1[M,*]@B1 (+ A2[M,*]@B2) + bias1 (+ bias2) )
// M must be a multiple of 16 (one block per 16 rows). K = kt*32 <= 384 and
// the A buffers are zero-padded to K, so staging needs no tail guards.
// The block's A tile is converted to bf16 in LDS once; fragments come from
// ds_load_b128 instead of 8x-redundant per-wave global loads + cvt chains.
// ---------------------------------------------------------------------------
__global__ __launch_bounds__(256)
void gemm_wmma_kernel(const float* __restrict__ A1, int lda1, int kt1,
                      const uint4* __restrict__ B1,
                      const float* __restrict__ A2, int lda2, int kt2,
                      const uint4* __restrict__ B2,
                      const float* __restrict__ bias1,
                      const float* __restrict__ bias2,
                      float* __restrict__ C, int ldc, int act) {
    __shared__ __bf16 s_a[16][QINP];   // 12 KB bf16 A tile stage

    const int lane = threadIdx.x & 31;
    const int wave = threadIdx.x >> 5;
    const int row0 = blockIdx.x * 16;

    v8f acc[2];
    acc[0] = v8f_zero();
    acc[1] = v8f_zero();

    // ---- stage + multiply phase 1 ----
    {
        const int kp = kt1 * 32;
        for (int m = 0; m < 16; ++m) {
            const float* src = A1 + (size_t)(row0 + m) * lda1;
            for (int k = threadIdx.x; k < kp; k += 256)
                s_a[m][k] = (__bf16)src[k];
        }
    }
    __syncthreads();
    for (int ks = 0; ks < kt1; ++ks) {
        FragU a;
        load_afrag_lds(&s_a[0][0], QINP, 0, ks * 32, lane, a);
        #pragma unroll
        for (int i = 0; i < 2; ++i) {
            FragU b;
            load_bfrag(B1, (wave * 2 + i) * kt1 + ks, lane, b);
            acc[i] = __builtin_amdgcn_wmma_f32_16x16x32_bf16(
                false, a.bf, false, b.bf, (short)0, acc[i], false, false);
        }
    }
    // ---- stage + multiply phase 2 (optional second A/B pair) ----
    if (A2 != nullptr) {
        __syncthreads();   // everyone done reading phase-1 stage
        {
            const int kp = kt2 * 32;
            for (int m = 0; m < 16; ++m) {
                const float* src = A2 + (size_t)(row0 + m) * lda2;
                for (int k = threadIdx.x; k < kp; k += 256)
                    s_a[m][k] = (__bf16)src[k];
            }
        }
        __syncthreads();
        for (int ks = 0; ks < kt2; ++ks) {
            FragU a;
            load_afrag_lds(&s_a[0][0], QINP, 0, ks * 32, lane, a);
            #pragma unroll
            for (int i = 0; i < 2; ++i) {
                FragU b;
                load_bfrag(B2, (wave * 2 + i) * kt2 + ks, lane, b);
                acc[i] = __builtin_amdgcn_wmma_f32_16x16x32_bf16(
                    false, a.bf, false, b.bf, (short)0, acc[i], false, false);
            }
        }
    }
    // ---- epilogue ----
    const int mb = (lane >> 4) * 8;
    #pragma unroll
    for (int i = 0; i < 2; ++i) {
        const int c = (wave * 2 + i) * 16 + (lane & 15);
        float bi = (bias1 != nullptr) ? bias1[c] : 0.0f;
        if (bias2 != nullptr) bi += bias2[c];
        #pragma unroll
        for (int v = 0; v < 8; ++v) {
            float val = acc[i][v] + bi;
            if (act) val = fmaxf(val, 0.0f);
            C[(size_t)(row0 + mb + v) * ldc + c] = val;
        }
    }
}

// ---------------------------------------------------------------------------
// GRU memory update: h_new = GRU(unique_msg, memory[unique_nids])
// One block = 16 node rows; 8 waves x 6 N-tiles = 48 tiles (768 gate cols).
// A tiles are staged once in LDS as bf16; the stage regions are ALIASED with
// the gate-staging buffers inside one 64 KB block:
//   bytes [0,32K)  : s_rz   (float)   | s_amsg (bf16 16x800, 25.6 KB)
//   bytes [32K,48K): s_gin  (float)
//   bytes [48K,64K): s_ghn  (float)   | s_ahh  (bf16 16x256, 8 KB)
// Ordering: fill stages -> sync -> gi WMMA (amsg) -> sync -> gi store
// (clobbers amsg region only) -> gh WMMA (ahh intact) -> sync -> gh store
// (clobbers ahh region) -> sync -> gate math.
// ---------------------------------------------------------------------------
__global__ __launch_bounds__(256)
void gru_wmma_kernel(const float* __restrict__ memory,
                     const float* __restrict__ unique_msg,
                     const float* __restrict__ b_ih,
                     const float* __restrict__ b_hh,
                     const int*   __restrict__ unique_nids,
                     const uint4* __restrict__ wih_pack,
                     const uint4* __restrict__ whh_pack,
                     float* __restrict__ h_ws) {
    __shared__ __align__(16) char smem[65536];
    float  (*s_rz)[512]  = (float (*)[512])(smem);
    float  (*s_gin)[256] = (float (*)[256])(smem + 32768);
    float  (*s_ghn)[256] = (float (*)[256])(smem + 49152);
    __bf16 (*s_amsg)[MSGP] = (__bf16 (*)[MSGP])(smem);          // aliases s_rz
    __bf16 (*s_ahh)[256]   = (__bf16 (*)[256])(smem + 49152);   // aliases s_ghn

    const int lane = threadIdx.x & 31;
    const int wave = threadIdx.x >> 5;
    const int row0 = blockIdx.x * 16;
    const int mb   = (lane >> 4) * 8;

    // ---- stage both A tiles (single f32->bf16 convert per element) ----
    for (int idx = threadIdx.x; idx < 16 * MSGP; idx += 256) {
        const int m = idx / MSGP;
        const int k = idx - m * MSGP;
        const float v = (k < MSGD) ? unique_msg[(size_t)(row0 + m) * MSGD + k] : 0.0f;
        s_amsg[m][k] = (__bf16)v;
    }
    for (int idx = threadIdx.x; idx < 16 * DMDIM; idx += 256) {
        const int m = idx >> 8;
        const int c = idx & 255;
        const int u = unique_nids[row0 + m];
        s_ahh[m][c] = (__bf16)memory[(size_t)u * DMDIM + c];
    }
    __syncthreads();

    v8f acc[6];
    // ---- gi = unique_msg @ w_ih^T ----
    #pragma unroll
    for (int i = 0; i < 6; ++i) acc[i] = v8f_zero();
    for (int ks = 0; ks < KT_IH; ++ks) {
        FragU a;
        load_afrag_lds(&s_amsg[0][0], MSGP, 0, ks * 32, lane, a);
        #pragma unroll
        for (int i = 0; i < 6; ++i) {
            FragU b;
            load_bfrag(wih_pack, (wave * 6 + i) * KT_IH + ks, lane, b);
            acc[i] = __builtin_amdgcn_wmma_f32_16x16x32_bf16(
                false, a.bf, false, b.bf, (short)0, acc[i], false, false);
        }
    }
    __syncthreads();   // all waves done reading s_amsg before gi store clobbers it
    #pragma unroll
    for (int i = 0; i < 6; ++i) {
        const int c  = (wave * 6 + i) * 16 + (lane & 15);
        const float bi = b_ih[c];
        #pragma unroll
        for (int v = 0; v < 8; ++v) {
            const float val = acc[i][v] + bi;
            const int   m   = mb + v;
            if (c < 512) s_rz[m][c]        = val;
            else         s_gin[m][c - 512] = val;
        }
    }
    // ---- gh = memory[unique_nids] @ w_hh^T (s_ahh still intact) ----
    #pragma unroll
    for (int i = 0; i < 6; ++i) acc[i] = v8f_zero();
    for (int ks = 0; ks < KT_HH; ++ks) {
        FragU a;
        load_afrag_lds(&s_ahh[0][0], DMDIM, 0, ks * 32, lane, a);
        #pragma unroll
        for (int i = 0; i < 6; ++i) {
            FragU b;
            load_bfrag(whh_pack, (wave * 6 + i) * KT_HH + ks, lane, b);
            acc[i] = __builtin_amdgcn_wmma_f32_16x16x32_bf16(
                false, a.bf, false, b.bf, (short)0, acc[i], false, false);
        }
    }
    __syncthreads();   // all waves done reading s_ahh before gh store clobbers it
    #pragma unroll
    for (int i = 0; i < 6; ++i) {
        const int c  = (wave * 6 + i) * 16 + (lane & 15);
        const float bh = b_hh[c];
        #pragma unroll
        for (int v = 0; v < 8; ++v) {
            const float val = acc[i][v] + bh;
            const int   m   = mb + v;
            if (c < 512) s_rz[m][c]       += val;
            else         s_ghn[m][c - 512] = val;
        }
    }
    __syncthreads();
    // ---- gate math ----
    for (int idx = threadIdx.x; idx < 16 * DMDIM; idx += 256) {
        const int m = idx >> 8;
        const int c = idx & 255;
        const int row = row0 + m;
        const float r  = 1.0f / (1.0f + __expf(-s_rz[m][c]));
        const float z  = 1.0f / (1.0f + __expf(-s_rz[m][c + 256]));
        const float nn = tanhf(s_gin[m][c] + r * s_ghn[m][c]);
        const int   u  = unique_nids[row];
        const float h  = memory[(size_t)u * DMDIM + c];
        h_ws[(size_t)row * DMDIM + c] = (1.0f - z) * nn + z * h;
    }
}

// ---------------------------------------------------------------------------
// Build q_in = [node_feat | cos(t2v_b) | 0pad] : [NBATCH, QINP] fp32.
// Columns [0,256) double as the gathered node features for the output layer.
// ---------------------------------------------------------------------------
__global__ void build_qin_kernel(const float* __restrict__ memory,
                                 const float* __restrict__ h_ws,
                                 const int*   __restrict__ inv,
                                 const int*   __restrict__ nids,
                                 const float* __restrict__ t2v_b,
                                 float* __restrict__ qin) {
    const int idx = blockIdx.x * blockDim.x + threadIdx.x;
    if (idx >= NBATCH * QINP) return;
    const int n = idx / QINP;
    const int c = idx - n * QINP;
    float val = 0.0f;
    if (c < DMDIM) {
        const int nid = nids[n];
        const int j   = inv[nid];
        val = (j >= 0) ? h_ws[(size_t)j * DMDIM + c]
                       : memory[(size_t)nid * DMDIM + c];
    } else if (c < QIND) {
        val = __cosf(t2v_b[c - DMDIM]);
    }
    qin[idx] = val;
}

// ---------------------------------------------------------------------------
// Temporal attention core. One block per batch node: K/V projections via
// WMMA from LDS-staged bf16 neighbor features; scores/softmax scalar (K=20).
// ---------------------------------------------------------------------------
__global__ __launch_bounds__(256)
void attn_wmma_kernel(const float* __restrict__ memory,
                      const float* __restrict__ last_update,
                      const float* __restrict__ times,
                      const float* __restrict__ nbr_times,
                      const float* __restrict__ nbr_feats,
                      const float* __restrict__ t2v_w,
                      const float* __restrict__ t2v_b,
                      const int*   __restrict__ nids,
                      const int*   __restrict__ nbr_nids,
                      const int*   __restrict__ nbr_mask,
                      const int*   __restrict__ inv,
                      const float* __restrict__ h_ws,
                      const float* __restrict__ q_buf,
                      const uint4* __restrict__ wk_pack,
                      const uint4* __restrict__ wv_pack,
                      float* __restrict__ ao_buf) {
    __shared__ __bf16 s_kin[32][KINP];   // padded neighbor features, bf16
    __shared__ float  s_kv[KNBR][DEMBD]; // K then V projections
    __shared__ float  s_q[DEMBD];
    __shared__ float  s_sc[2][KNBR];
    __shared__ float  s_attn[2][KNBR];
    __shared__ float  s_tadj;

    const int n    = blockIdx.x;
    const int tid  = threadIdx.x;
    const int lane = tid & 31;
    const int wave = tid >> 5;

    if (tid == 0) {
        const int nid = nids[n];
        s_tadj = times[n] - last_update[nid];
    }
    for (int c = tid; c < DEMBD; c += 256) s_q[c] = q_buf[(size_t)n * DEMBD + c];
    __syncthreads();
    const float tadj = s_tadj;

    // ---- build k_in = [nbr_memory | nbr_feats | time2vec(delta)] as bf16 ----
    for (int idx = tid; idx < 32 * KINP; idx += 256) {
        const int j = idx / KINP;
        const int c = idx - j * KINP;
        float val = 0.0f;
        if (j < KNBR && c < KIN) {
            if (c < DMDIM) {
                const int r  = nbr_nids[n * KNBR + j];
                const int jj = inv[r];
                val = (jj >= 0) ? h_ws[(size_t)jj * DMDIM + c]
                                : memory[(size_t)r * DMDIM + c];
            } else if (c < DMDIM + DEDIM) {
                val = nbr_feats[((size_t)n * KNBR + j) * DEDIM + (c - DMDIM)];
            } else {
                const int d = c - (DMDIM + DEDIM);
                const float delta = nbr_times[n * KNBR + j] - tadj;
                val = __cosf(delta * t2v_w[d] + t2v_b[d]);
            }
        }
        s_kin[j][c] = (__bf16)val;
    }
    __syncthreads();

    // ---- K projection via WMMA: wave -> (mtile = wave&1, 4 ntiles) ----
    const int mtile = wave & 1;
    const int nt0   = (wave >> 1) * 4;
    {
        v8f acc[4];
        #pragma unroll
        for (int i = 0; i < 4; ++i) acc[i] = v8f_zero();
        for (int ks = 0; ks < KT_KV; ++ks) {
            FragU a;
            load_afrag_lds(&s_kin[0][0], KINP, mtile * 16, ks * 32, lane, a);
            #pragma unroll
            for (int i = 0; i < 4; ++i) {
                FragU b;
                load_bfrag(wk_pack, (nt0 + i) * KT_KV + ks, lane, b);
                acc[i] = __builtin_amdgcn_wmma_f32_16x16x32_bf16(
                    false, a.bf, false, b.bf, (short)0, acc[i], false, false);
            }
        }
        const int mb = mtile * 16 + (lane >> 4) * 8;
        #pragma unroll
        for (int i = 0; i < 4; ++i) {
            const int c = (nt0 + i) * 16 + (lane & 15);
            #pragma unroll
            for (int v = 0; v < 8; ++v) {
                const int m = mb + v;
                if (m < KNBR) s_kv[m][c] = acc[i][v];
            }
        }
    }
    __syncthreads();

    // ---- scores + mask ----
    if (tid < 2 * KNBR) {
        const int h = tid / KNBR, j = tid - (tid / KNBR) * KNBR;
        float sc = 0.0f;
        for (int d = 0; d < 128; ++d) sc += s_q[h * 128 + d] * s_kv[j][h * 128 + d];
        sc *= 0.08838834764831845f;   // 1/sqrt(128)
        if (nbr_mask[n * KNBR + j] <= 0) sc = -1e9f;
        s_sc[h][j] = sc;
    }
    __syncthreads();
    if (tid < 2) {
        const int h = tid;
        float mx = -1e30f;
        for (int j = 0; j < KNBR; ++j) mx = fmaxf(mx, s_sc[h][j]);
        float e[KNBR], sum = 0.0f;
        for (int j = 0; j < KNBR; ++j) { e[j] = __expf(s_sc[h][j] - mx); sum += e[j]; }
        const float is = 1.0f / sum;
        for (int j = 0; j < KNBR; ++j) s_attn[h][j] = e[j] * is;
    }
    __syncthreads();

    // ---- V projection via WMMA (overwrites s_kv) ----
    {
        v8f acc[4];
        #pragma unroll
        for (int i = 0; i < 4; ++i) acc[i] = v8f_zero();
        for (int ks = 0; ks < KT_KV; ++ks) {
            FragU a;
            load_afrag_lds(&s_kin[0][0], KINP, mtile * 16, ks * 32, lane, a);
            #pragma unroll
            for (int i = 0; i < 4; ++i) {
                FragU b;
                load_bfrag(wv_pack, (nt0 + i) * KT_KV + ks, lane, b);
                acc[i] = __builtin_amdgcn_wmma_f32_16x16x32_bf16(
                    false, a.bf, false, b.bf, (short)0, acc[i], false, false);
            }
        }
        const int mb = mtile * 16 + (lane >> 4) * 8;
        #pragma unroll
        for (int i = 0; i < 4; ++i) {
            const int c = (nt0 + i) * 16 + (lane & 15);
            #pragma unroll
            for (int v = 0; v < 8; ++v) {
                const int m = mb + v;
                if (m < KNBR) s_kv[m][c] = acc[i][v];
            }
        }
    }
    __syncthreads();

    // ---- attn_out -> global ----
    for (int c = tid; c < DEMBD; c += 256) {
        const int h = c >> 7;
        float ao = 0.0f;
        for (int j = 0; j < KNBR; ++j) ao += s_attn[h][j] * s_kv[j][c];
        ao_buf[(size_t)n * DEMBD + c] = ao;
    }
}

// ---------------------------------------------------------------------------
// Final link-predictor output: out[b] = sigmoid(hh[b]·lp_out_w + lp_out_b)
// ---------------------------------------------------------------------------
__global__ __launch_bounds__(256)
void lp_out_kernel(const float* __restrict__ hh,
                   const float* __restrict__ ow,
                   const float* __restrict__ ob,
                   float* __restrict__ out) {
    __shared__ float s_red[DEMBD];
    const int b   = blockIdx.x;
    const int tid = threadIdx.x;
    s_red[tid] = hh[(size_t)b * DEMBD + tid] * ow[tid];
    __syncthreads();
    for (int s = 128; s > 0; s >>= 1) {
        if (tid < s) s_red[tid] += s_red[tid + s];
        __syncthreads();
    }
    if (tid == 0) out[b] = 1.0f / (1.0f + __expf(-(s_red[0] + ob[0])));
}

// ---------------------------------------------------------------------------
// Launch
// ---------------------------------------------------------------------------
extern "C" void kernel_launch(void* const* d_in, const int* in_sizes, int n_in,
                              void* d_out, int out_size, void* d_ws, size_t ws_size,
                              hipStream_t stream) {
    const float* memory      = (const float*)d_in[0];
    const float* last_update = (const float*)d_in[1];
    const float* unique_msg  = (const float*)d_in[2];
    const float* times       = (const float*)d_in[3];
    const float* nbr_times   = (const float*)d_in[4];
    const float* nbr_feats   = (const float*)d_in[5];
    const float* t2v_w       = (const float*)d_in[6];
    const float* t2v_b       = (const float*)d_in[7];
    const float* gru_w_ih    = (const float*)d_in[8];
    const float* gru_w_hh    = (const float*)d_in[9];
    const float* gru_b_ih    = (const float*)d_in[10];
    const float* gru_b_hh    = (const float*)d_in[11];
    const float* w_q         = (const float*)d_in[12];
    const float* w_k         = (const float*)d_in[13];
    const float* w_v         = (const float*)d_in[14];
    const float* w_out       = (const float*)d_in[15];
    const float* b_out       = (const float*)d_in[16];
    const float* lp_src_w    = (const float*)d_in[17];
    const float* lp_src_b    = (const float*)d_in[18];
    const float* lp_dst_w    = (const float*)d_in[19];
    const float* lp_dst_b    = (const float*)d_in[20];
    const float* lp_out_w    = (const float*)d_in[21];
    const float* lp_out_b    = (const float*)d_in[22];
    const int*   unique_nids = (const int*)d_in[23];
    const int*   nids        = (const int*)d_in[24];
    const int*   nbr_nids    = (const int*)d_in[25];
    const int*   nbr_mask    = (const int*)d_in[26];
    float* out = (float*)d_out;

    // Workspace layout
    char* ws = (char*)d_ws;
    size_t off = 0;
    float* h_ws   = (float*)(ws + off); off += (size_t)UU * DMDIM * 4;        // 51.2 MB
    int*   inv    = (int*)(ws + off);   off += (size_t)NNODE * 4;             // 0.4 MB
    float* qin    = (float*)(ws + off); off += (size_t)NBATCH * QINP * 4;     // 9.2 MB
    float* q_buf  = (float*)(ws + off); off += (size_t)NBATCH * DEMBD * 4;    // 6.1 MB
    float* ao_buf = (float*)(ws + off); off += (size_t)NBATCH * DEMBD * 4;    // 6.1 MB
    float* z_buf  = (float*)(ws + off); off += (size_t)NBATCH * DEMBD * 4;    // 6.1 MB
    float* hh_buf = (float*)(ws + off); off += (size_t)4000 * DEMBD * 4;      // 4.1 MB
    off = (off + 255) & ~(size_t)255;
    uint32_t* wih_pack   = (uint32_t*)(ws + off); off += (size_t)NT_G   * KT_IH * 1024;
    uint32_t* whh_pack   = (uint32_t*)(ws + off); off += (size_t)NT_G   * KT_HH * 1024;
    uint32_t* wk_pack    = (uint32_t*)(ws + off); off += (size_t)NT_256 * KT_KV * 1024;
    uint32_t* wv_pack    = (uint32_t*)(ws + off); off += (size_t)NT_256 * KT_KV * 1024;
    uint32_t* wq_pack    = (uint32_t*)(ws + off); off += (size_t)NT_256 * KT_Q  * 1024;
    uint32_t* wout0_pack = (uint32_t*)(ws + off); off += (size_t)NT_256 * KT_256 * 1024;
    uint32_t* wout1_pack = (uint32_t*)(ws + off); off += (size_t)NT_256 * KT_256 * 1024;
    uint32_t* wsw_pack   = (uint32_t*)(ws + off); off += (size_t)NT_256 * KT_256 * 1024;
    uint32_t* wdw_pack   = (uint32_t*)(ws + off); off += (size_t)NT_256 * KT_256 * 1024;

    init_inv_kernel<<<(NNODE + 255) / 256, 256, 0, stream>>>(inv, NNODE);
    fill_inv_kernel<<<(UU + 255) / 256, 256, 0, stream>>>(inv, unique_nids, UU);

    // Weight packs: (W, out, Nout, Kin, ld, col0, NT, KT)
    pack_wmma_b<<<NT_G * KT_IH, 256, 0, stream>>>(gru_w_ih, wih_pack, G3, MSGD, MSGD, 0, NT_G, KT_IH);
    pack_wmma_b<<<NT_G * KT_HH, 256, 0, stream>>>(gru_w_hh, whh_pack, G3, DMDIM, DMDIM, 0, NT_G, KT_HH);
    pack_wmma_b<<<NT_256 * KT_KV, 256, 0, stream>>>(w_k, wk_pack, DEMBD, KIN, KIN, 0, NT_256, KT_KV);
    pack_wmma_b<<<NT_256 * KT_KV, 256, 0, stream>>>(w_v, wv_pack, DEMBD, KIN, KIN, 0, NT_256, KT_KV);
    pack_wmma_b<<<NT_256 * KT_Q, 256, 0, stream>>>(w_q, wq_pack, DEMBD, QIND, QIND, 0, NT_256, KT_Q);
    pack_wmma_b<<<NT_256 * KT_256, 256, 0, stream>>>(w_out, wout0_pack, DEMBD, DEMBD, 512, 0, NT_256, KT_256);
    pack_wmma_b<<<NT_256 * KT_256, 256, 0, stream>>>(w_out, wout1_pack, DEMBD, DMDIM, 512, 256, NT_256, KT_256);
    pack_wmma_b<<<NT_256 * KT_256, 256, 0, stream>>>(lp_src_w, wsw_pack, DEMBD, DEMBD, DEMBD, 0, NT_256, KT_256);
    pack_wmma_b<<<NT_256 * KT_256, 256, 0, stream>>>(lp_dst_w, wdw_pack, DEMBD, DEMBD, DEMBD, 0, NT_256, KT_256);

    // GRU memory update
    gru_wmma_kernel<<<UU / 16, 256, 0, stream>>>(
        memory, unique_msg, gru_b_ih, gru_b_hh, unique_nids,
        (const uint4*)wih_pack, (const uint4*)whh_pack, h_ws);

    // q_in gather, then q = q_in @ w_q^T (WMMA)
    build_qin_kernel<<<(NBATCH * QINP + 255) / 256, 256, 0, stream>>>(
        memory, h_ws, inv, nids, t2v_b, qin);
    gemm_wmma_kernel<<<NBATCH / 16, 256, 0, stream>>>(
        qin, QINP, KT_Q, (const uint4*)wq_pack,
        nullptr, 0, 0, nullptr, nullptr, nullptr, q_buf, DEMBD, 0);

    // attention core -> attn_out
    attn_wmma_kernel<<<NBATCH, 256, 0, stream>>>(
        memory, last_update, times, nbr_times, nbr_feats, t2v_w, t2v_b,
        nids, nbr_nids, nbr_mask, inv, h_ws, q_buf,
        (const uint4*)wk_pack, (const uint4*)wv_pack, ao_buf);

    // z = relu(attn_out @ Wo[:, :256] + node_feat @ Wo[:, 256:] + b_out)
    gemm_wmma_kernel<<<NBATCH / 16, 256, 0, stream>>>(
        ao_buf, DEMBD, KT_256, (const uint4*)wout0_pack,
        qin, QINP, KT_256, (const uint4*)wout1_pack,
        b_out, nullptr, z_buf, DEMBD, 1);

    // link predictor hidden layers: pos(src,dst) and neg(src,neg)
    gemm_wmma_kernel<<<2000 / 16, 256, 0, stream>>>(
        z_buf, DEMBD, KT_256, (const uint4*)wsw_pack,
        z_buf + (size_t)2000 * DEMBD, DEMBD, KT_256, (const uint4*)wdw_pack,
        lp_src_b, lp_dst_b, hh_buf, DEMBD, 1);
    gemm_wmma_kernel<<<2000 / 16, 256, 0, stream>>>(
        z_buf, DEMBD, KT_256, (const uint4*)wsw_pack,
        z_buf + (size_t)4000 * DEMBD, DEMBD, KT_256, (const uint4*)wdw_pack,
        lp_src_b, lp_dst_b, hh_buf + (size_t)2000 * DEMBD, DEMBD, 1);

    // out = sigmoid(hh @ lp_out_w^T + lp_out_b), pos rows then neg rows
    lp_out_kernel<<<4000, 256, 0, stream>>>(hh_buf, lp_out_w, lp_out_b, out);
}
